// PatientGNN_19078244729027
// MI455X (gfx1250) — matrix operand order repaired and checked by reading, not compiled, not run on hardware
//
#include <hip/hip_runtime.h>
#include <hip/hip_bf16.h>

// ---------------------------------------------------------------------------
// PatientGNN on MI455X (gfx1250): GCN -> ReLU -> GAT -> ReLU -> GCN
// Dense blocks via v_wmma_f32_16x16x32_bf16 (wave32, one wave per 16-row
// stripe holding all n-tile accumulators), sparse blocks via L2-resident
// wave-per-edge float atomics.
// ---------------------------------------------------------------------------

#define GNN_N      50000
#define GNN_E      800000
#define GNN_INC    128
#define GNN_HID    32
#define GNN_OUTC   64
#define GNN_HEADS  4
#define NEG_SLOPE  0.2f

typedef __attribute__((ext_vector_type(16))) __bf16          bf16x16;
typedef __attribute__((ext_vector_type(8)))  float           f32x8;
typedef __attribute__((ext_vector_type(8)))  unsigned short  u16x8;
typedef __attribute__((ext_vector_type(16))) unsigned short  u16x16;

union BF16Cast16 { u16x16 u; bf16x16 b; };

__device__ __forceinline__ unsigned short f32_to_bf16(float f) {
    unsigned int u = __float_as_uint(f);
    u += 0x7FFFu + ((u >> 16) & 1u);           // round-to-nearest-even
    return (unsigned short)(u >> 16);
}

__device__ __forceinline__ float leaky(float x) {
    return x > 0.0f ? x : NEG_SLOPE * x;
}

// ---------------------------------------------------------------------------
// Degree / normalization
// ---------------------------------------------------------------------------
__global__ void k_fill_f32(float* __restrict__ p, float v, int n) {
    int i = blockIdx.x * blockDim.x + threadIdx.x;
    if (i < n) p[i] = v;
}

__global__ void k_degree(const long long* __restrict__ ei, float* __restrict__ deg, int e) {
    int i = blockIdx.x * blockDim.x + threadIdx.x;
    if (i < e) atomicAdd(&deg[(size_t)ei[(size_t)GNN_E + i]], 1.0f);
}

__global__ void k_rsqrt(const float* __restrict__ deg, float* __restrict__ dinv, int n) {
    int i = blockIdx.x * blockDim.x + threadIdx.x;
    if (i < n) dinv[i] = __frsqrt_rn(deg[i]);
}

// ---------------------------------------------------------------------------
// fp32 -> bf16 conversion (plain / fused ReLU)
// ---------------------------------------------------------------------------
__global__ void k_cvt_bf16(const float* __restrict__ in, unsigned short* __restrict__ out, int n) {
    int i = blockIdx.x * blockDim.x + threadIdx.x;
    if (i < n) out[i] = f32_to_bf16(in[i]);
}

__global__ void k_relu_bf16(const float* __restrict__ in, unsigned short* __restrict__ out, int n) {
    int i = blockIdx.x * blockDim.x + threadIdx.x;
    if (i < n) out[i] = f32_to_bf16(fmaxf(in[i], 0.0f));
}

// ---------------------------------------------------------------------------
// Pack weight matrix W[K x Nn] (row-major fp32) into the per-lane WMMA
// B-matrix layout: column lane&15; e<8 -> K = (lane>>4)*8+e, e>=8 ->
// K = 16+(lane>>4)*8+(e-8)  (per 32-deep K tile). 16 contiguous ushorts per lane.
// ---------------------------------------------------------------------------
__global__ void k_pack_w(const float* __restrict__ W, unsigned short* __restrict__ P,
                         int K, int Nn) {
    int total = (K >> 5) * (Nn >> 4) * 32 * 16;
    int idx = blockIdx.x * blockDim.x + threadIdx.x;
    if (idx >= total) return;
    int e    = idx & 15;
    int lane = (idx >> 4) & 31;
    int tile = idx >> 9;
    int ntiles = Nn >> 4;
    int kt = tile / ntiles;
    int nt = tile - kt * ntiles;
    int k   = (kt << 5) + ((e >> 3) << 4) + ((lane >> 4) << 3) + (e & 7);
    int col = (nt << 4) + (lane & 15);
    P[idx] = f32_to_bf16(W[(size_t)k * Nn + col]);
}

// ---------------------------------------------------------------------------
// WMMA GEMM: C[M x NN] (f32) = A[M x K] (bf16 row-major) @ Bpacked.
// One wave per 16-ROW STRIPE: the A fragment is loaded once per K-step and
// reused across all NN/16 n-tile accumulators (independent D registers ->
// no WMMA->WMMA RAW chains, A VMEM traffic cut by NN/16).
// A per-lane layout (16-bit A 16x32): row = lane&15,
//   elements 0..7  = K + (lane>>4)*8 .. +7          (one 16B load)
//   elements 8..15 = K + 16 + (lane>>4)*8 .. +7     (one 16B load)
// ---------------------------------------------------------------------------
template <int K, int NN>
__global__ void k_wmma_gemm(const unsigned short* __restrict__ A,
                            const unsigned short* __restrict__ Bp,
                            float* __restrict__ C, int M) {
    constexpr int NT = NN / 16;
    constexpr int KT = K / 32;
    int wid  = (blockIdx.x * blockDim.x + threadIdx.x) >> 5;
    int lane = threadIdx.x & 31;
    if (wid >= (M >> 4)) return;              // wave = one 16-row stripe
    int half = lane >> 4;
    int l15  = lane & 15;
    int row  = (wid << 4) + l15;

    f32x8 acc[NT];
#pragma unroll
    for (int nt = 0; nt < NT; ++nt) acc[nt] = f32x8{};

#pragma unroll
    for (int kt = 0; kt < KT; ++kt) {
        const unsigned short* ap = A + (size_t)row * K + (kt << 5) + (half << 3);
        u16x8 alo = *(const u16x8*)ap;          // 16B: K .. K+7
        u16x8 ahi = *(const u16x8*)(ap + 16);   // 16B: K+16 .. K+23
        BF16Cast16 av;
#pragma unroll
        for (int i = 0; i < 8; ++i) { av.u[i] = alo[i]; av.u[8 + i] = ahi[i]; }
#pragma unroll
        for (int nt = 0; nt < NT; ++nt) {
            BF16Cast16 bv;
            bv.u = *(const u16x16*)(Bp + ((size_t)(kt * NT + nt) * 32 + lane) * 16);
            acc[nt] = __builtin_amdgcn_wmma_f32_16x16x32_bf16(
                        false, av.b, false, bv.b, (short)0, acc[nt], false, false);
        }
    }
    // C layout: lane holds column l15, rows half*8 .. half*8+7
    int rbase = (wid << 4) + (half << 3);
#pragma unroll
    for (int nt = 0; nt < NT; ++nt) {
        int col = (nt << 4) + l15;
#pragma unroll
        for (int r = 0; r < 8; ++r)
            C[(size_t)(rbase + r) * NN + col] = acc[nt][r];
    }
}

// ---------------------------------------------------------------------------
// GCN scatter: init with self-loop term + bias, then wave-per-edge atomics.
// ---------------------------------------------------------------------------
__global__ void k_gcn_self(const float* __restrict__ pre, const float* __restrict__ dinv,
                           const float* __restrict__ bias, float* __restrict__ acc,
                           int n, int F) {
    int i = blockIdx.x * blockDim.x + threadIdx.x;
    if (i >= n * F) return;
    int node = i / F, c = i - node * F;
    float dv = dinv[node];
    acc[i] = pre[i] * dv * dv + bias[c];
}

template <int F>
__global__ void k_gcn_edge(const long long* __restrict__ ei,
                           const float* __restrict__ pre, const float* __restrict__ dinv,
                           float* __restrict__ acc, int e) {
    int w = (blockIdx.x * blockDim.x + threadIdx.x) >> 5;
    int lane = threadIdx.x & 31;
    if (w >= e) return;
    long long s = ei[w];
    long long d = ei[(size_t)GNN_E + w];
    float norm = dinv[s] * dinv[d];
#pragma unroll
    for (int c0 = 0; c0 < F; c0 += 32) {
        int c = c0 + lane;
        atomicAdd(&acc[(size_t)d * F + c], pre[(size_t)s * F + c] * norm);
    }
}

// ---------------------------------------------------------------------------
// GAT
// ---------------------------------------------------------------------------
// thread per (node, head): attention coefficients + self-logit seeds the max
__global__ void k_att_coef(const float* __restrict__ hg,
                           const float* __restrict__ ags, const float* __restrict__ agd,
                           float* __restrict__ als, float* __restrict__ ald,
                           float* __restrict__ m, int n) {
    int t = blockIdx.x * blockDim.x + threadIdx.x;
    if (t >= n * GNN_HEADS) return;
    int node = t >> 2, h = t & 3;
    const float* hp = hg + (size_t)node * (GNN_HEADS * GNN_HID) + h * GNN_HID;
    float s = 0.0f, d = 0.0f;
#pragma unroll
    for (int k = 0; k < GNN_HID; ++k) {
        s += hp[k] * ags[h * GNN_HID + k];
        d += hp[k] * agd[h * GNN_HID + k];
    }
    als[t] = s;
    ald[t] = d;
    m[t] = leaky(s + d);          // self-loop logit initializes segment max
}

__global__ void k_att_max(const long long* __restrict__ ei,
                          const float* __restrict__ als, const float* __restrict__ ald,
                          float* __restrict__ m, int e) {
    int i = blockIdx.x * blockDim.x + threadIdx.x;
    if (i >= e) return;
    long long s = ei[i], d = ei[(size_t)GNN_E + i];
#pragma unroll
    for (int h = 0; h < GNN_HEADS; ++h)
        atomicMax(&m[(size_t)d * 4 + h], leaky(als[(size_t)s * 4 + h] + ald[(size_t)d * 4 + h]));
}

__global__ void k_den_init(const float* __restrict__ als, const float* __restrict__ ald,
                           const float* __restrict__ m, float* __restrict__ den, int n) {
    int t = blockIdx.x * blockDim.x + threadIdx.x;
    if (t >= n * GNN_HEADS) return;
    den[t] = __expf(leaky(als[t] + ald[t]) - m[t]);   // self-loop term
}

__global__ void k_den_edge(const long long* __restrict__ ei,
                           const float* __restrict__ als, const float* __restrict__ ald,
                           const float* __restrict__ m, float* __restrict__ den, int e) {
    int i = blockIdx.x * blockDim.x + threadIdx.x;
    if (i >= e) return;
    long long s = ei[i], d = ei[(size_t)GNN_E + i];
#pragma unroll
    for (int h = 0; h < GNN_HEADS; ++h) {
        float l = leaky(als[(size_t)s * 4 + h] + ald[(size_t)d * 4 + h]);
        atomicAdd(&den[(size_t)d * 4 + h], __expf(l - m[(size_t)d * 4 + h]));
    }
}

// self-loop message: accG[n, h*32+c] = hg * alpha_self
__global__ void k_gat_self(const float* __restrict__ hg,
                           const float* __restrict__ als, const float* __restrict__ ald,
                           const float* __restrict__ m, const float* __restrict__ den,
                           float* __restrict__ accG, int n) {
    int t = blockIdx.x * blockDim.x + threadIdx.x;
    if (t >= n * GNN_HEADS * GNN_HID) return;
    int node = t >> 7;
    int h = (t >> 5) & 3;
    size_t nh = (size_t)node * 4 + h;
    float alpha = __expf(leaky(als[nh] + ald[nh]) - m[nh]) / den[nh];
    accG[t] = hg[t] * alpha;
}

// wave per edge: 4 heads x 32 features
__global__ void k_gat_edge(const long long* __restrict__ ei, const float* __restrict__ hg,
                           const float* __restrict__ als, const float* __restrict__ ald,
                           const float* __restrict__ m, const float* __restrict__ den,
                           float* __restrict__ accG, int e) {
    int w = (blockIdx.x * blockDim.x + threadIdx.x) >> 5;
    int lane = threadIdx.x & 31;
    if (w >= e) return;
    long long s = ei[w], d = ei[(size_t)GNN_E + w];
#pragma unroll
    for (int h = 0; h < GNN_HEADS; ++h) {
        float l = leaky(als[(size_t)s * 4 + h] + ald[(size_t)d * 4 + h]);
        float alpha = __expf(l - m[(size_t)d * 4 + h]) / den[(size_t)d * 4 + h];
        atomicAdd(&accG[(size_t)d * 128 + h * 32 + lane],
                  hg[(size_t)s * 128 + h * 32 + lane] * alpha);
    }
}

// mean over heads + bias + ReLU -> bf16 for next GEMM
__global__ void k_gat_fin(const float* __restrict__ accG, const float* __restrict__ bg,
                          unsigned short* __restrict__ outb, int n) {
    int t = blockIdx.x * blockDim.x + threadIdx.x;
    if (t >= n * GNN_HID) return;
    int node = t >> 5, c = t & 31;
    const float* p = accG + (size_t)node * 128;
    float v = 0.25f * (p[c] + p[32 + c] + p[64 + c] + p[96 + c]) + bg[c];
    outb[t] = f32_to_bf16(fmaxf(v, 0.0f));
}

// ---------------------------------------------------------------------------
// Host-side launcher
// ---------------------------------------------------------------------------
extern "C" void kernel_launch(void* const* d_in, const int* in_sizes, int n_in,
                              void* d_out, int out_size, void* d_ws, size_t ws_size,
                              hipStream_t stream) {
    (void)in_sizes; (void)n_in; (void)out_size; (void)ws_size;
    const float*     x    = (const float*)d_in[0];
    const long long* ei   = (const long long*)d_in[1];   // int64 [2, E]
    const float*     W1   = (const float*)d_in[2];
    const float*     b1   = (const float*)d_in[3];
    const float*     Wg   = (const float*)d_in[4];
    const float*     ags  = (const float*)d_in[5];
    const float*     agd  = (const float*)d_in[6];
    const float*     bg   = (const float*)d_in[7];
    const float*     W2   = (const float*)d_in[8];
    const float*     b2   = (const float*)d_in[9];
    float* out = (float*)d_out;

    // workspace carve-up (256B aligned slabs)
    char* w = (char*)d_ws;
    auto alloc = [&](size_t bytes) -> void* {
        void* p = (void*)w;
        w += (bytes + 255) & ~(size_t)255;
        return p;
    };
    const int N = GNN_N, E = GNN_E;
    float*          deg    = (float*)alloc((size_t)N * 4);
    float*          dinv   = (float*)alloc((size_t)N * 4);
    unsigned short* xb     = (unsigned short*)alloc((size_t)N * GNN_INC * 2);
    unsigned short* w1p    = (unsigned short*)alloc((size_t)(GNN_INC/32)*(GNN_HID/16)*512 * 2);
    float*          h1pre  = (float*)alloc((size_t)N * GNN_HID * 4);
    float*          acc1   = (float*)alloc((size_t)N * GNN_HID * 4);
    unsigned short* h1b    = (unsigned short*)alloc((size_t)N * GNN_HID * 2);
    unsigned short* wgp    = (unsigned short*)alloc((size_t)(GNN_HID/32)*(128/16)*512 * 2);
    float*          hg     = (float*)alloc((size_t)N * 128 * 4);
    float*          als    = (float*)alloc((size_t)N * 4 * 4);
    float*          ald    = (float*)alloc((size_t)N * 4 * 4);
    float*          mmax   = (float*)alloc((size_t)N * 4 * 4);
    float*          den    = (float*)alloc((size_t)N * 4 * 4);
    float*          accG   = (float*)alloc((size_t)N * 128 * 4);
    unsigned short* hcb    = (unsigned short*)alloc((size_t)N * GNN_HID * 2);
    unsigned short* w2p    = (unsigned short*)alloc((size_t)(GNN_HID/32)*(GNN_OUTC/16)*512 * 2);
    float*          h2pre  = (float*)alloc((size_t)N * GNN_OUTC * 4);

    const int B = 256;
    auto g = [](long long total, int blk) { return (int)((total + blk - 1) / blk); };
    const int gemm_threads = (N / 16) * 32;   // one wave per 16-row stripe

    // --- degrees / symmetric norm (self-loop => init deg = 1) ---
    k_fill_f32<<<g(N, B), B, 0, stream>>>(deg, 1.0f, N);
    k_degree <<<g(E, B), B, 0, stream>>>(ei, deg, E);
    k_rsqrt  <<<g(N, B), B, 0, stream>>>(deg, dinv, N);

    // --- GCN1: h1pre = x @ W1 (bf16 WMMA) ---
    k_cvt_bf16<<<g((long long)N * GNN_INC, B), B, 0, stream>>>(x, xb, N * GNN_INC);
    k_pack_w  <<<g((GNN_INC/32)*(GNN_HID/16)*512, B), B, 0, stream>>>(W1, w1p, GNN_INC, GNN_HID);
    k_wmma_gemm<GNN_INC, GNN_HID><<<g(gemm_threads, B), B, 0, stream>>>(xb, w1p, h1pre, N);
    k_gcn_self<<<g((long long)N * GNN_HID, B), B, 0, stream>>>(h1pre, dinv, b1, acc1, N, GNN_HID);
    k_gcn_edge<GNN_HID><<<g((long long)E * 32, B), B, 0, stream>>>(ei, h1pre, dinv, acc1, E);
    k_relu_bf16<<<g((long long)N * GNN_HID, B), B, 0, stream>>>(acc1, h1b, N * GNN_HID);

    // --- GAT: hg = h1 @ Wg (bf16 WMMA), segment softmax, message pass ---
    k_pack_w<<<g((GNN_HID/32)*(128/16)*512, B), B, 0, stream>>>(Wg, wgp, GNN_HID, 128);
    k_wmma_gemm<GNN_HID, 128><<<g(gemm_threads, B), B, 0, stream>>>(h1b, wgp, hg, N);
    k_att_coef<<<g((long long)N * 4, B), B, 0, stream>>>(hg, ags, agd, als, ald, mmax, N);
    k_att_max <<<g(E, B), B, 0, stream>>>(ei, als, ald, mmax, E);
    k_den_init<<<g((long long)N * 4, B), B, 0, stream>>>(als, ald, mmax, den, N);
    k_den_edge<<<g(E, B), B, 0, stream>>>(ei, als, ald, mmax, den, E);
    k_gat_self<<<g((long long)N * 128, B), B, 0, stream>>>(hg, als, ald, mmax, den, accG, N);
    k_gat_edge<<<g((long long)E * 32, B), B, 0, stream>>>(ei, hg, als, ald, mmax, den, accG, E);
    k_gat_fin <<<g((long long)N * GNN_HID, B), B, 0, stream>>>(accG, bg, hcb, N);

    // --- GCN2: h2pre = h @ W2 (bf16 WMMA), scatter straight into d_out ---
    k_pack_w<<<g((GNN_HID/32)*(GNN_OUTC/16)*512, B), B, 0, stream>>>(W2, w2p, GNN_HID, GNN_OUTC);
    k_wmma_gemm<GNN_HID, GNN_OUTC><<<g(gemm_threads, B), B, 0, stream>>>(hcb, w2p, h2pre, N);
    k_gcn_self<<<g((long long)N * GNN_OUTC, B), B, 0, stream>>>(h2pre, dinv, b2, out, N, GNN_OUTC);
    k_gcn_edge<GNN_OUTC><<<g((long long)E * 32, B), B, 0, stream>>>(ei, h2pre, dinv, out, E);
}
